// PCN_encoder22_25744033972951
// MI455X (gfx1250) — matrix operand order repaired and checked by reading, compiled, not run on hardware
//
#include <hip/hip_runtime.h>
#include <hip/hip_bf16.h>

typedef __attribute__((ext_vector_type(16))) _Float16 v16h;
typedef __attribute__((ext_vector_type(8)))  _Float16 v8h;
typedef __attribute__((ext_vector_type(8)))  float    v8f;

#define TPB 256

// ---------------------------------------------------------------------------
// [B,3,N] -> [B,N,3]
__global__ void k_transpose_xyz(const float* __restrict__ in, float* __restrict__ out,
                                int N, int Btot) {
  size_t idx = (size_t)blockIdx.x * blockDim.x + threadIdx.x;
  size_t total = (size_t)Btot * N * 3;
  if (idx >= total) return;
  int b = (int)(idx / ((size_t)N * 3));
  int rem = (int)(idx % ((size_t)N * 3));
  int n = rem / 3, c = rem % 3;
  out[idx] = in[((size_t)b * 3 + c) * N + n];
}

// ---------------------------------------------------------------------------
// Farthest point sampling: one block per batch. dist kept in LDS.
__global__ __launch_bounds__(TPB)
void k_fps(const float* __restrict__ xyz, int* __restrict__ fidx, int N, int npoint) {
  __shared__ float dist[2048];
  __shared__ float rval[TPB];
  __shared__ int   ridx[TPB];
  int b = blockIdx.x, tid = threadIdx.x;
  const float* p = xyz + (size_t)b * N * 3;
  for (int i = tid; i < N; i += TPB) dist[i] = 1e10f;
  int far = 0;
  if (tid == 0) fidx[(size_t)b * npoint] = 0;
  __syncthreads();
  for (int it = 1; it < npoint; ++it) {
    float cx = p[far * 3 + 0], cy = p[far * 3 + 1], cz = p[far * 3 + 2];
    float best = -1.0f; int bi = 0x7fffffff;
    for (int i = tid; i < N; i += TPB) {
      float dx = p[i * 3 + 0] - cx;
      float dy = p[i * 3 + 1] - cy;
      float dz = p[i * 3 + 2] - cz;
      float d = dx * dx + dy * dy + dz * dz;
      float m = fminf(dist[i], d);
      dist[i] = m;
      if (m > best) { best = m; bi = i; }   // keeps first max (i increasing)
    }
    rval[tid] = best; ridx[tid] = bi;
    __syncthreads();
    for (int off = TPB / 2; off > 0; off >>= 1) {
      if (tid < off) {
        float ov = rval[tid + off]; int oi = ridx[tid + off];
        if (ov > rval[tid] || (ov == rval[tid] && oi < ridx[tid])) {
          rval[tid] = ov; ridx[tid] = oi;
        }
      }
      __syncthreads();
    }
    far = ridx[0];
    if (tid == 0) fidx[(size_t)b * npoint + it] = far;
    __syncthreads();
  }
}

// ---------------------------------------------------------------------------
// new_xyz[b,s,:] = xyz[b, fidx[b,s], :]
__global__ void k_gather_xyz(const float* __restrict__ xyz, const int* __restrict__ fidx,
                             float* __restrict__ out, int N, int S, int Btot) {
  size_t idx = (size_t)blockIdx.x * blockDim.x + threadIdx.x;
  size_t total = (size_t)Btot * S * 3;
  if (idx >= total) return;
  int b = (int)(idx / ((size_t)S * 3));
  int rem = (int)(idx % ((size_t)S * 3));
  int s = rem / 3, c = rem % 3;
  out[idx] = xyz[((size_t)b * N + fidx[(size_t)b * S + s]) * 3 + c];
}

// ---------------------------------------------------------------------------
// Ball query: one wave32 per (b,s); ballot-compacted ordered selection of the
// first `nsample` indices with d2<=r2; remainder filled with the first index.
__global__ __launch_bounds__(32)
void k_ball_query(const float* __restrict__ xyz, const float* __restrict__ nxyz,
                  int* __restrict__ gidx, int N, int S, int nsample, float r2) {
  int bs = blockIdx.x;                // b*S + s
  int lane = threadIdx.x & 31;
  int b = bs / S;
  const float* p = xyz + (size_t)b * N * 3;
  float qx = nxyz[(size_t)bs * 3 + 0];
  float qy = nxyz[(size_t)bs * 3 + 1];
  float qz = nxyz[(size_t)bs * 3 + 2];
  int* out = gidx + (size_t)bs * nsample;
  int count = 0, first = 0;
  bool haveFirst = false;
  for (int base = 0; base < N && count < nsample; base += 32) {
    int j = base + lane;              // N is a multiple of 32
    float dx = p[j * 3 + 0] - qx;
    float dy = p[j * 3 + 1] - qy;
    float dz = p[j * 3 + 2] - qz;
    float d = dx * dx + dy * dy + dz * dz;
    bool pred = (d <= r2);
    unsigned int mask = (unsigned int)__ballot(pred);
    int prefix = __popc(mask & ((1u << lane) - 1u));
    int slot = count + prefix;
    if (pred && slot < nsample) out[slot] = j;
    if (!haveFirst && mask) { first = base + __ffs(mask) - 1; haveFirst = true; }
    count += __popc(mask);
  }
  for (int t = count + lane; t < nsample; t += 32) out[t] = first;
}

// ---------------------------------------------------------------------------
// feats[row, c] (f16, row = (b*S+s)*K + j):
//   c<3      : xyz[b, gidx[row], c] - new_xyz[b, s, c]
//   c<3+D    : pts[b, gidx[row], c-3]
//   else     : 0 (pad to Kpad)
__global__ void k_build_group_feats(const float* __restrict__ xyz, const float* __restrict__ nxyz,
                                    const float* __restrict__ pts, const int* __restrict__ gidx,
                                    _Float16* __restrict__ f, int N, int S, int K, int D,
                                    int Kpad, int Btot) {
  size_t idx = (size_t)blockIdx.x * blockDim.x + threadIdx.x;
  size_t total = (size_t)Btot * S * K * Kpad;
  if (idx >= total) return;
  size_t row = idx / Kpad;
  int c = (int)(idx % Kpad);
  int b = (int)(row / ((size_t)S * K));
  int rem = (int)(row % ((size_t)S * K));
  int s = rem / K;
  float v = 0.0f;
  if (c < 3) {
    int pi = gidx[row];
    v = xyz[((size_t)b * N + pi) * 3 + c] - nxyz[((size_t)b * S + s) * 3 + c];
  } else if (c < 3 + D) {
    int pi = gidx[row];
    v = pts[((size_t)b * N + pi) * D + (c - 3)];
  }
  f[idx] = (_Float16)v;
}

// group_all: feats[row=b*S+s, c] = c<3 ? xyzS[row,c] : (c<3+D ? pts[row,c-3] : 0)
__global__ void k_build_concat(const float* __restrict__ xyzS, const float* __restrict__ pts,
                               _Float16* __restrict__ f, int S, int D, int Kpad, int Btot) {
  size_t idx = (size_t)blockIdx.x * blockDim.x + threadIdx.x;
  size_t total = (size_t)Btot * S * Kpad;
  if (idx >= total) return;
  size_t row = idx / Kpad;
  int c = (int)(idx % Kpad);
  float v = 0.0f;
  if (c < 3) v = xyzS[row * 3 + c];
  else if (c < 3 + D) v = pts[row * (size_t)D + (c - 3)];
  f[idx] = (_Float16)v;
}

// ---------------------------------------------------------------------------
// W[Cin,Cout] f32 -> WT[Cout,Kpad] f16 (transposed, zero-padded K)
__global__ void k_prep_wt(const float* __restrict__ W, _Float16* __restrict__ WT,
                          int Cin, int Cout, int Kpad) {
  size_t idx = (size_t)blockIdx.x * blockDim.x + threadIdx.x;
  size_t total = (size_t)Cout * Kpad;
  if (idx >= total) return;
  int n = (int)(idx / Kpad);
  int k = (int)(idx % Kpad);
  WT[idx] = (k < Cin) ? (_Float16)W[(size_t)k * Cout + n] : (_Float16)0.0f;
}

// ---------------------------------------------------------------------------
// WMMA GEMM: C[M,Nout] f32 = A[M,K] f16 x BT[Nout,K]^T f16.
// One wave computes a 32x64 tile (2 M-subtiles x 4 N-subtiles): each B
// fragment is shared by two WMMAs, each A fragment by four -> ~22 FLOP/B
// from L2. Fragment layouts per CDNA5 ISA 16-bit A (16x32) / B (32x16)
// wave32 striping.
__global__ __launch_bounds__(32)
void k_wmma_gemm(const _Float16* __restrict__ A, const _Float16* __restrict__ BT,
                 float* __restrict__ C, int M, int K, int Nout) {
  const int lane = threadIdx.x & 31;
  const int half = lane >> 4;     // 0: lanes 0-15, 1: lanes 16-31
  const int l15  = lane & 15;
  const int tM = blockIdx.x << 5;
  const int tN = blockIdx.y << 6;
  v8f acc0[4] = {v8f{}, v8f{}, v8f{}, v8f{}};
  v8f acc1[4] = {v8f{}, v8f{}, v8f{}, v8f{}};
  const _Float16* arow0 = A + (size_t)(tM + l15) * K + half * 8;
  const _Float16* arow1 = A + (size_t)(tM + 16 + l15) * K + half * 8;
  const _Float16* bbase = BT + (size_t)(tN + l15) * K + half * 8;
  for (int k0 = 0; k0 < K; k0 += 32) {
    // A 16x32 f16: lane(l15) row M; halves 0..7 -> K=k0+8h..+7, 8..15 -> +16
    v8h a0lo = *(const v8h*)(arow0 + k0);
    v8h a0hi = *(const v8h*)(arow0 + k0 + 16);
    v8h a1lo = *(const v8h*)(arow1 + k0);
    v8h a1hi = *(const v8h*)(arow1 + k0 + 16);
    v16h a0, a1;
#pragma unroll
    for (int i = 0; i < 8; ++i) {
      a0[i] = a0lo[i]; a0[i + 8] = a0hi[i];
      a1[i] = a1lo[i]; a1[i + 8] = a1hi[i];
    }
#pragma unroll
    for (int t = 0; t < 4; ++t) {
      // B 32x16 f16: lane(l15) col N; halves 0..7 -> K=k0+8h..+7, 8..15 -> +16
      const _Float16* br = bbase + (size_t)(t * 16) * K + k0;
      v8h blo = *(const v8h*)br;
      v8h bhi = *(const v8h*)(br + 16);
      v16h bb;
#pragma unroll
      for (int i = 0; i < 8; ++i) { bb[i] = blo[i]; bb[i + 8] = bhi[i]; }
      acc0[t] = __builtin_amdgcn_wmma_f32_16x16x32_f16(
          false, a0, false, bb, (short)0, acc0[t], false, false);
      acc1[t] = __builtin_amdgcn_wmma_f32_16x16x32_f16(
          false, a1, false, bb, (short)0, acc1[t], false, false);
    }
  }
  // C/D layout: VGPR r holds row (subtile base) + r + 8*half, col = tN + t*16 + l15
#pragma unroll
  for (int t = 0; t < 4; ++t) {
    float* crow0 = C + (size_t)(tM + half * 8) * Nout + tN + t * 16 + l15;
    float* crow1 = C + (size_t)(tM + 16 + half * 8) * Nout + tN + t * 16 + l15;
#pragma unroll
    for (int r = 0; r < 8; ++r) {
      crow0[(size_t)r * Nout] = acc0[t][r];
      crow1[(size_t)r * Nout] = acc1[t][r];
    }
  }
}

// ---------------------------------------------------------------------------
// Per-channel batch statistics over Mrows rows (population var, as jnp.var)
__global__ __launch_bounds__(TPB)
void k_bn_stats(const float* __restrict__ x, float* __restrict__ mean,
                float* __restrict__ rstd, int Mrows, int C) {
  __shared__ float s1[TPB], s2[TPB];
  int c = blockIdx.x, tid = threadIdx.x;
  float a = 0.0f, b2 = 0.0f;
  for (int i = tid; i < Mrows; i += TPB) {
    float v = x[(size_t)i * C + c];
    a += v; b2 += v * v;
  }
  s1[tid] = a; s2[tid] = b2;
  __syncthreads();
  for (int off = TPB / 2; off > 0; off >>= 1) {
    if (tid < off) { s1[tid] += s1[tid + off]; s2[tid] += s2[tid + off]; }
    __syncthreads();
  }
  if (tid == 0) {
    float m = s1[0] / (float)Mrows;
    float var = s2[0] / (float)Mrows - m * m;
    mean[c] = m;
    rstd[c] = rsqrtf(var + 1e-5f);
  }
}

// BN(+gamma,beta)+ReLU. If nxt != null, also emits f16 [Mrows, KpadNext]
// (zero-padded) for the next GEMM; else normalizes x in place.
__global__ void k_bn_apply(float* __restrict__ x, const float* __restrict__ mean,
                           const float* __restrict__ rstd, const float* __restrict__ g,
                           const float* __restrict__ be, _Float16* __restrict__ nxt,
                           int Mrows, int C, int KpadNext) {
  size_t idx = (size_t)blockIdx.x * blockDim.x + threadIdx.x;
  if (nxt) {
    size_t total = (size_t)Mrows * KpadNext;
    if (idx >= total) return;
    size_t row = idx / KpadNext;
    int c = (int)(idx % KpadNext);
    if (c < C) {
      float v = x[row * (size_t)C + c];
      v = fmaxf((v - mean[c]) * rstd[c] * g[c] + be[c], 0.0f);
      nxt[idx] = (_Float16)v;
    } else {
      nxt[idx] = (_Float16)0.0f;
    }
  } else {
    size_t total = (size_t)Mrows * C;
    if (idx >= total) return;
    int c = (int)(idx % C);
    float v = x[idx];
    x[idx] = fmaxf((v - mean[c]) * rstd[c] * g[c] + be[c], 0.0f);
  }
}

// max over the K (group) axis: x[BS, K, C] -> out[BS, C]
__global__ void k_maxpool(const float* __restrict__ x, float* __restrict__ out,
                          int BS, int K, int C) {
  size_t idx = (size_t)blockIdx.x * blockDim.x + threadIdx.x;
  size_t total = (size_t)BS * C;
  if (idx >= total) return;
  size_t bs = idx / C;
  int c = (int)(idx % C);
  float m = -3.402823466e38f;
  for (int k = 0; k < K; ++k)
    m = fmaxf(m, x[((size_t)bs * K + k) * C + c]);
  out[idx] = m;
}

// ---------------------------------------------------------------------------
static inline unsigned cdiv(size_t a, size_t b) { return (unsigned)((a + b - 1) / b); }

extern "C" void kernel_launch(void* const* d_in, const int* in_sizes, int n_in,
                              void* d_out, int out_size, void* d_ws, size_t ws_size,
                              hipStream_t stream) {
  (void)in_sizes; (void)n_in; (void)out_size; (void)ws_size;
  const int B = 8, N0 = 2048;
  const int S1 = 512, K1 = 32;       // SA1 grouping
  const int S2 = 128, K2 = 64;       // SA2 grouping
  const int M1 = B * S1 * K1;        // 131072
  const int M2 = B * S2 * K2;        // 65536
  const int M3 = B * S2;             // 1024
  const float r1sq = 0.0176f * 0.0176f;
  const float r2sq = 2.3466f * 2.3466f;

  const float* xyz_in = (const float*)d_in[0];

  // bump allocator over d_ws
  char* wsb = (char*)d_ws;
  size_t off = 0;
  auto alloc = [&](size_t bytes) -> void* {
    off = (off + 255) & ~(size_t)255;
    void* p = wsb + off;
    off += bytes;
    return p;
  };
  float*    xyzT  = (float*)alloc((size_t)B * N0 * 3 * 4);
  int*      fidx1 = (int*)alloc((size_t)B * S1 * 4);
  float*    nxyz1 = (float*)alloc((size_t)B * S1 * 3 * 4);
  int*      gidx1 = (int*)alloc((size_t)B * S1 * K1 * 4);
  int*      fidx2 = (int*)alloc((size_t)B * S2 * 4);
  float*    nxyz2 = (float*)alloc((size_t)B * S2 * 3 * 4);
  int*      gidx2 = (int*)alloc((size_t)B * S2 * K2 * 4);
  float*    l1pts = (float*)alloc((size_t)B * S1 * 256 * 4);
  float*    l2pts = (float*)alloc((size_t)B * S2 * 1024 * 4);
  float*    meanb = (float*)alloc(1024 * 4);
  float*    rstdb = (float*)alloc(1024 * 4);
  _Float16* wt0 = (_Float16*)alloc((size_t)64   * 32   * 2);
  _Float16* wt1 = (_Float16*)alloc((size_t)128  * 64   * 2);
  _Float16* wt2 = (_Float16*)alloc((size_t)256  * 128  * 2);
  _Float16* wt3 = (_Float16*)alloc((size_t)256  * 288  * 2);
  _Float16* wt4 = (_Float16*)alloc((size_t)512  * 256  * 2);
  _Float16* wt5 = (_Float16*)alloc((size_t)1024 * 512  * 2);
  _Float16* wt6 = (_Float16*)alloc((size_t)1024 * 1056 * 2);
  _Float16* wt7 = (_Float16*)alloc((size_t)1024 * 1024 * 2);
  _Float16* f16a = (_Float16*)alloc((size_t)64 * 1024 * 1024);
  _Float16* f16b = (_Float16*)alloc((size_t)64 * 1024 * 1024);
  float*    f32buf = (float*)alloc((size_t)M2 * 1024 * 4);   // 256 MB, max activation

  // one MLP layer: prep weights -> WMMA GEMM -> BN stats -> BN+ReLU (+f16 repack)
  auto run_layer = [&](const _Float16* fin, _Float16* fnext, int Mrows, int Kpad,
                       int Cin, int Cout, int KpadNext,
                       const float* Wp, const float* g, const float* be, _Float16* wtb) {
    k_prep_wt<<<cdiv((size_t)Cout * Kpad, TPB), TPB, 0, stream>>>(Wp, wtb, Cin, Cout, Kpad);
    k_wmma_gemm<<<dim3(Mrows / 32, Cout / 64), dim3(32), 0, stream>>>(
        fin, wtb, f32buf, Mrows, Kpad, Cout);
    k_bn_stats<<<dim3(Cout), dim3(TPB), 0, stream>>>(f32buf, meanb, rstdb, Mrows, Cout);
    size_t total = (size_t)Mrows * (fnext ? KpadNext : Cout);
    k_bn_apply<<<cdiv(total, TPB), TPB, 0, stream>>>(
        f32buf, meanb, rstdb, g, be, fnext, Mrows, Cout, KpadNext);
  };

  // ---- input transpose ----
  k_transpose_xyz<<<cdiv((size_t)B * N0 * 3, TPB), TPB, 0, stream>>>(xyz_in, xyzT, N0, B);

  // ---- SA1 ----
  k_fps<<<dim3(B), dim3(TPB), 0, stream>>>(xyzT, fidx1, N0, S1);
  k_gather_xyz<<<cdiv((size_t)B * S1 * 3, TPB), TPB, 0, stream>>>(xyzT, fidx1, nxyz1, N0, S1, B);
  k_ball_query<<<dim3(B * S1), dim3(32), 0, stream>>>(xyzT, nxyz1, gidx1, N0, S1, K1, r1sq);
  k_build_group_feats<<<cdiv((size_t)M1 * 32, TPB), TPB, 0, stream>>>(
      xyzT, nxyz1, nullptr, gidx1, f16a, N0, S1, K1, 0, 32, B);
  run_layer(f16a, f16b, M1, 32, 3, 64, 64,
            (const float*)d_in[1], (const float*)d_in[3], (const float*)d_in[4], wt0);
  run_layer(f16b, f16a, M1, 64, 64, 128, 128,
            (const float*)d_in[5], (const float*)d_in[7], (const float*)d_in[8], wt1);
  run_layer(f16a, nullptr, M1, 128, 128, 256, 0,
            (const float*)d_in[9], (const float*)d_in[11], (const float*)d_in[12], wt2);
  k_maxpool<<<cdiv((size_t)B * S1 * 256, TPB), TPB, 0, stream>>>(f32buf, l1pts, B * S1, K1, 256);

  // ---- SA2 ----
  k_fps<<<dim3(B), dim3(TPB), 0, stream>>>(nxyz1, fidx2, S1, S2);
  k_gather_xyz<<<cdiv((size_t)B * S2 * 3, TPB), TPB, 0, stream>>>(nxyz1, fidx2, nxyz2, S1, S2, B);
  k_ball_query<<<dim3(B * S2), dim3(32), 0, stream>>>(nxyz1, nxyz2, gidx2, S1, S2, K2, r2sq);
  k_build_group_feats<<<cdiv((size_t)M2 * 288, TPB), TPB, 0, stream>>>(
      nxyz1, nxyz2, l1pts, gidx2, f16a, S1, S2, K2, 256, 288, B);
  run_layer(f16a, f16b, M2, 288, 259, 256, 256,
            (const float*)d_in[13], (const float*)d_in[15], (const float*)d_in[16], wt3);
  run_layer(f16b, f16a, M2, 256, 256, 512, 512,
            (const float*)d_in[17], (const float*)d_in[19], (const float*)d_in[20], wt4);
  run_layer(f16a, nullptr, M2, 512, 512, 1024, 0,
            (const float*)d_in[21], (const float*)d_in[23], (const float*)d_in[24], wt5);
  k_maxpool<<<cdiv((size_t)B * S2 * 1024, TPB), TPB, 0, stream>>>(f32buf, l2pts, B * S2, K2, 1024);

  // ---- SA3 (group_all) ----
  k_build_concat<<<cdiv((size_t)M3 * 1056, TPB), TPB, 0, stream>>>(
      nxyz2, l2pts, f16a, S2, 1024, 1056, B);
  run_layer(f16a, f16b, M3, 1056, 1027, 1024, 1024,
            (const float*)d_in[25], (const float*)d_in[27], (const float*)d_in[28], wt6);
  run_layer(f16b, nullptr, M3, 1024, 1024, 1024, 0,
            (const float*)d_in[29], (const float*)d_in[31], (const float*)d_in[32], wt7);
  k_maxpool<<<cdiv((size_t)B * 1024, TPB), TPB, 0, stream>>>(f32buf, (float*)d_out, B, S2, 1024);
}
